// RWKV7TMix_18219251270214
// MI455X (gfx1250) — matrix operand order repaired and checked by reading, not compiled, not run on hardware
//
#include <hip/hip_runtime.h>
#include <math.h>

// ---------------------------------------------------------------------------
// RWKV7 time-mix for MI455X (gfx1250, wave32, WMMA).
// GEMMs: v_wmma_f32_16x16x32_f16, f32 accumulate. All weight matrices are
// converted (and transposed where needed) to f16 N-major-by-K layout so every
// A and B fragment is two 16B loads.
// Wave tile = (16*MT) x (16*NT); explicit double-buffered K pipeline so the
// next k-step's global loads stay outstanding under the current WMMAs.
// ---------------------------------------------------------------------------

typedef __attribute__((ext_vector_type(16))) _Float16 v16h;
typedef __attribute__((ext_vector_type(8)))  _Float16 v8h;
typedef __attribute__((ext_vector_type(8)))  float    v8f;

#define BB   2
#define TT   1024
#define CC   2048
#define HH   32
#define NN   64
#define MM   (BB*TT)          // 2048 rows
#define MC   ((size_t)MM*CC)  // 4M elements

// ---------------------------------------------------------------------------
// Elementwise helpers
// ---------------------------------------------------------------------------
__device__ __forceinline__ float sigm(float x) { return 1.0f / (1.0f + __expf(-x)); }

__device__ __forceinline__ float wave_sum32(float v) {
#pragma unroll
  for (int off = 16; off > 0; off >>= 1) v += __shfl_xor(v, off, 32);
  return v;
}

// ---------------------------------------------------------------------------
// Weight conversion kernels (run each launch; deterministic)
// ---------------------------------------------------------------------------
__global__ __launch_bounds__(256) void cvt_f16(const float* __restrict__ in,
                                               _Float16* __restrict__ out, int n) {
  int i = blockIdx.x * 256 + threadIdx.x;
  if (i < n) out[i] = (_Float16)in[i];
}

// in is [K][N] row-major; out is [N][K] row-major (f16)
__global__ __launch_bounds__(256) void cvt_f16_t(const float* __restrict__ in,
                                                 _Float16* __restrict__ out, int K, int N) {
  int i = blockIdx.x * 256 + threadIdx.x;
  if (i < K * N) {
    int k = i / N, n = i % N;
    out[(size_t)n * K + k] = (_Float16)in[i];
  }
}

// ---------------------------------------------------------------------------
// Prep: time-shift + six mixes, f32 -> f16 activation streams
// ---------------------------------------------------------------------------
__global__ __launch_bounds__(256) void prep_mix(
    const float* __restrict__ x,
    const float* __restrict__ mr, const float* __restrict__ mw,
    const float* __restrict__ mk, const float* __restrict__ mv,
    const float* __restrict__ ma, const float* __restrict__ mg,
    _Float16* __restrict__ xr, _Float16* __restrict__ xw,
    _Float16* __restrict__ xk, _Float16* __restrict__ xv,
    _Float16* __restrict__ xa, _Float16* __restrict__ xg) {
  size_t idx = (size_t)blockIdx.x * 256 + threadIdx.x;   // < M*C
  int c = (int)(idx & (CC - 1));
  size_t row = idx >> 11;                                // /C
  int t = (int)(row & (TT - 1));
  float xc = x[idx];
  float prev = (t == 0) ? 0.0f : x[idx - CC];
  float xx = prev - xc;
  xr[idx] = (_Float16)(xc + xx * mr[c]);
  xw[idx] = (_Float16)(xc + xx * mw[c]);
  xk[idx] = (_Float16)(xc + xx * mk[c]);
  xv[idx] = (_Float16)(xc + xx * mv[c]);
  xa[idx] = (_Float16)(xc + xx * ma[c]);
  xg[idx] = (_Float16)(xc + xx * mg[c]);
}

// ---------------------------------------------------------------------------
// WMMA GEMM: Out[M,N] = A[M,K] (f16, row-major) * Bnk[N,K]^T (f16, row-major)
// One wave -> (16*MT) x (16*NT) tile. Block = 8 waves. Grids divide exactly so
// EXEC is all-ones (WMMA requirement). f32 accumulate, fused epilogue.
// Double-buffered K pipeline: issue loads for k+32 before computing k.
// MODE: 0 f32 id | 1 f16 id | 2 f16 tanh | 3 f32 decay(bias) |
//       4 f32 sigmoid(bias) | 5 f16 sigmoid(bias)
// ---------------------------------------------------------------------------
template <int MT, int NT, int MODE>
__global__ __launch_bounds__(256) void gemm_wmma(
    const _Float16* __restrict__ A, const _Float16* __restrict__ Bnk,
    float* __restrict__ outF, _Float16* __restrict__ outH,
    const float* __restrict__ bias, int M, int N, int K) {
  const int lane = threadIdx.x & 31;
  const int wave = blockIdx.x * 8 + (threadIdx.x >> 5);
  const int nWaveTiles = N / (16 * NT);
  const int mt = wave / nWaveTiles;
  const int nt = wave % nWaveTiles;
  const int m0 = mt * 16 * MT;
  const int n0 = nt * 16 * NT;

  const int lm = lane & 15;          // row within A frag / col within B,C frags
  const int hi = lane >> 4;          // lane-half select
  const int kbA = hi << 3;           // A: K sub-base 0 / 8
  const int kbB = hi << 3;           // B: K sub-base 0 / 8

  const _Float16* Abase = A + (size_t)(m0 + lm) * K + kbA;
  const _Float16* Bbase = Bnk + (size_t)(n0 + lm) * K + kbB;

  v8f acc[MT][NT];
#pragma unroll
  for (int u = 0; u < MT; ++u)
#pragma unroll
    for (int t = 0; t < NT; ++t)
      acc[u][t] = (v8f){0.f, 0.f, 0.f, 0.f, 0.f, 0.f, 0.f, 0.f};

  // Two fragment sets for software pipelining.
  v16h afrag[2][MT];
  v16h bfrag[2][NT];

  auto load_frags = [&](int buf, int k0) {
#pragma unroll
    for (int u = 0; u < MT; ++u) {
      const _Float16* Arow = Abase + (size_t)(u * 16) * K + k0;
      v8h alo = *(const v8h*)(Arow);
      v8h ahi = *(const v8h*)(Arow + 16);
#pragma unroll
      for (int i = 0; i < 8; ++i) {
        afrag[buf][u][i] = alo[i];
        afrag[buf][u][i + 8] = ahi[i];
      }
    }
#pragma unroll
    for (int t = 0; t < NT; ++t) {
      const _Float16* Brow = Bbase + (size_t)(t * 16) * K + k0;
      v8h blo = *(const v8h*)(Brow);
      v8h bhi = *(const v8h*)(Brow + 16);
#pragma unroll
      for (int i = 0; i < 8; ++i) {
        bfrag[buf][t][i] = blo[i];
        bfrag[buf][t][i + 8] = bhi[i];
      }
    }
  };

  load_frags(0, 0);
  int cur = 0;
  for (int k0 = 0; k0 < K; k0 += 32) {
    const int nk = k0 + 32;
    if (nk < K) load_frags(cur ^ 1, nk);   // uniform branch; loads fly under WMMAs
#pragma unroll
    for (int t = 0; t < NT; ++t)
#pragma unroll
      for (int u = 0; u < MT; ++u)
        acc[u][t] = __builtin_amdgcn_wmma_f32_16x16x32_f16(
            false, afrag[cur][u], false, bfrag[cur][t], (short)0, acc[u][t],
            false, false);
    cur ^= 1;
  }

  // Epilogue: C/D layout — VGPR r holds M = r + 8*hi, N = lane&15 (+16*t)
#pragma unroll
  for (int u = 0; u < MT; ++u) {
    const int rbase = m0 + u * 16 + (hi << 3);
#pragma unroll
    for (int t = 0; t < NT; ++t) {
      const int n = n0 + t * 16 + lm;
      float bv = 0.0f;
      if (MODE == 3 || MODE == 4 || MODE == 5) bv = bias ? bias[n] : 0.0f;
#pragma unroll
      for (int r = 0; r < 8; ++r) {
        const size_t off = (size_t)(rbase + r) * N + n;
        const float val = acc[u][t][r];
        if (MODE == 0) {
          outF[off] = val;
        } else if (MODE == 1) {
          outH[off] = (_Float16)val;
        } else if (MODE == 2) {
          outH[off] = (_Float16)tanhf(val);
        } else if (MODE == 3) {
          // dec = exp(-exp(-softplus(-(w0+z)) - 0.5)) = exp(-e^{-0.5}*sigmoid(w0+z))
          outF[off] = __expf(-0.60653065971263342f * sigm(bv + val));
        } else if (MODE == 4) {
          outF[off] = sigm(bv + val);
        } else {
          outH[off] = (_Float16)sigm(bv + val);
        }
      }
    }
  }
}

// ---------------------------------------------------------------------------
// Per-head pre-scan stage: kk normalization, aa/bb, k & v updates.
// One wave per (row m, head h); lane handles n=lane and n=lane+32.
// ---------------------------------------------------------------------------
__global__ __launch_bounds__(256) void head_mix(
    const float* __restrict__ ain, const float* __restrict__ svin,
    const float* __restrict__ vfirst,
    float* __restrict__ kio,           // in: k raw, out: k final (in place)
    float* __restrict__ vio,           // in: v raw, out: v mixed (in place)
    float* __restrict__ aaout, float* __restrict__ bbout,
    const float* __restrict__ k_k, const float* __restrict__ k_a) {
  const int g = blockIdx.x * 8 + (threadIdx.x >> 5);
  const int lane = threadIdx.x & 31;
  const int m = g >> 5, h = g & 31;
  const size_t base = (size_t)m * CC + h * NN;
  const int c0 = h * NN + lane, c1 = c0 + 32;
  const size_t i0 = base + lane, i1 = i0 + 32;

  float k0 = kio[i0], k1 = kio[i1];
  float kk0 = k0 * k_k[c0], kk1 = k1 * k_k[c1];
  float ss = wave_sum32(kk0 * kk0 + kk1 * kk1);
  float inv = 1.0f / fmaxf(sqrtf(ss), 1e-12f);
  float a0 = ain[i0], a1 = ain[i1];

  aaout[i0] = -kk0 * inv;       aaout[i1] = -kk1 * inv;
  bbout[i0] = kk0 * inv * a0;   bbout[i1] = kk1 * inv * a1;
  kio[i0] = k0 * (1.0f + (a0 - 1.0f) * k_a[c0]);
  kio[i1] = k1 * (1.0f + (a1 - 1.0f) * k_a[c1]);

  float v0 = vio[i0], v1 = vio[i1];
  vio[i0] = v0 + (vfirst[i0] - v0) * svin[i0];
  vio[i1] = v1 + (vfirst[i1] - v1) * svin[i1];
}

// ---------------------------------------------------------------------------
// WKV7 scan: one block (64 threads) per (b,h). Thread i owns state row
// S[i][0..63] in 64 VGPRs; per-step vectors broadcast via LDS.
// ---------------------------------------------------------------------------
__global__ __launch_bounds__(64) void wkv_scan(
    const float* __restrict__ r, const float* __restrict__ dec,
    const float* __restrict__ k, const float* __restrict__ v,
    const float* __restrict__ aa, const float* __restrict__ bb,
    float* __restrict__ o) {
  const int bh = blockIdx.x;
  const int b = bh >> 5, h = bh & 31;
  const int i = threadIdx.x;
  __shared__ float sA[NN], sB[NN], sK[NN], sR[NN], sD[NN];

  float S[NN];
#pragma unroll
  for (int j = 0; j < NN; ++j) S[j] = 0.0f;

  const size_t base = (size_t)b * TT * CC + (size_t)h * NN;
  for (int t = 0; t < TT; ++t) {
    const size_t idx = base + (size_t)t * CC + i;
    float vi = v[idx];
    sA[i] = aa[idx]; sB[i] = bb[idx]; sK[i] = k[idx];
    sR[i] = r[idx];  sD[i] = dec[idx];
    if (t + 1 < TT) {   // pull next step toward the WGP while we compute
      __builtin_prefetch((const void*)(r + idx + CC), 0, 1);
      __builtin_prefetch((const void*)(k + idx + CC), 0, 1);
      __builtin_prefetch((const void*)(dec + idx + CC), 0, 1);
    }
    __syncthreads();
    float sa = 0.0f;
#pragma unroll
    for (int j = 0; j < NN; ++j) sa += S[j] * sA[j];
    float out = 0.0f;
#pragma unroll
    for (int j = 0; j < NN; ++j) {
      S[j] = S[j] * sD[j] + sa * sB[j] + vi * sK[j];
      out += S[j] * sR[j];
    }
    o[idx] = out;
    __syncthreads();
  }
}

// ---------------------------------------------------------------------------
// Per-head post-scan: GroupNorm (per head over N), residual, gate -> f16
// ---------------------------------------------------------------------------
__global__ __launch_bounds__(256) void post_mix(
    const float* __restrict__ o, const float* __restrict__ r,
    const float* __restrict__ kf, const float* __restrict__ v,
    const float* __restrict__ g, const float* __restrict__ r_k,
    const float* __restrict__ ln_w, const float* __restrict__ ln_b,
    _Float16* __restrict__ og) {
  const int gi = blockIdx.x * 8 + (threadIdx.x >> 5);
  const int lane = threadIdx.x & 31;
  const int m = gi >> 5, h = gi & 31;
  const size_t base = (size_t)m * CC + h * NN;
  const int c0 = h * NN + lane, c1 = c0 + 32;
  const size_t i0 = base + lane, i1 = i0 + 32;

  float o0 = o[i0], o1 = o[i1];
  float mu = wave_sum32(o0 + o1) * (1.0f / NN);
  float d0 = o0 - mu, d1 = o1 - mu;
  float var = wave_sum32(d0 * d0 + d1 * d1) * (1.0f / NN);
  float is = rsqrtf(var + 6.4e-4f);  // eps * head_size_divisor^2 = 1e-5 * 64
  float n0 = d0 * is * ln_w[c0] + ln_b[c0];
  float n1 = d1 * is * ln_w[c1] + ln_b[c1];
  float dot = wave_sum32(r[i0] * kf[i0] * r_k[c0] + r[i1] * kf[i1] * r_k[c1]);
  og[i0] = (_Float16)((n0 + dot * v[i0]) * g[i0]);
  og[i1] = (_Float16)((n1 + dot * v[i1]) * g[i1]);
}

// ---------------------------------------------------------------------------
// Host launch
// ---------------------------------------------------------------------------
extern "C" void kernel_launch(void* const* d_in, const int* in_sizes, int n_in,
                              void* d_out, int out_size, void* d_ws, size_t ws_size,
                              hipStream_t stream) {
  (void)in_sizes; (void)n_in; (void)out_size; (void)ws_size;
  const float* x       = (const float*)d_in[0];
  const float* v_first = (const float*)d_in[1];
  const float* x_r = (const float*)d_in[2];
  const float* x_w = (const float*)d_in[3];
  const float* x_k = (const float*)d_in[4];
  const float* x_v = (const float*)d_in[5];
  const float* x_a = (const float*)d_in[6];
  const float* x_g = (const float*)d_in[7];
  const float* w0  = (const float*)d_in[8];
  const float* w1  = (const float*)d_in[9];
  const float* w2  = (const float*)d_in[10];
  const float* a0  = (const float*)d_in[11];
  const float* a1  = (const float*)d_in[12];
  const float* a2  = (const float*)d_in[13];
  const float* v0  = (const float*)d_in[14];
  const float* v1  = (const float*)d_in[15];
  const float* v2  = (const float*)d_in[16];
  const float* g1  = (const float*)d_in[17];
  const float* g2  = (const float*)d_in[18];
  const float* k_k = (const float*)d_in[19];
  const float* k_a = (const float*)d_in[20];
  const float* r_k = (const float*)d_in[21];
  const float* W_r = (const float*)d_in[22];
  const float* W_k = (const float*)d_in[23];
  const float* W_v = (const float*)d_in[24];
  const float* W_o = (const float*)d_in[25];
  const float* ln_w = (const float*)d_in[26];
  const float* ln_b = (const float*)d_in[27];
  float* outp = (float*)d_out;

  // ---- workspace carve-out -------------------------------------------------
  char* p = (char*)d_ws;
  auto alloc = [&](size_t bytes) -> void* {
    void* r = (void*)p;
    p += (bytes + 255) & ~(size_t)255;
    return r;
  };
  const size_t H2 = MC * sizeof(_Float16);
  const size_t F4 = MC * sizeof(float);
  _Float16 *xrh = (_Float16*)alloc(H2), *xwh = (_Float16*)alloc(H2),
           *xkh = (_Float16*)alloc(H2), *xvh = (_Float16*)alloc(H2),
           *xah = (_Float16*)alloc(H2), *xgh = (_Float16*)alloc(H2),
           *ogh = (_Float16*)alloc(H2);
  _Float16 *Wrh = (_Float16*)alloc(H2), *Wkh = (_Float16*)alloc(H2),
           *Wvh = (_Float16*)alloc(H2), *Woh = (_Float16*)alloc(H2);
  _Float16 *w1t = (_Float16*)alloc((size_t)96 * CC * 2),
           *w2t = (_Float16*)alloc((size_t)96 * CC * 2),
           *a1t = (_Float16*)alloc((size_t)96 * CC * 2),
           *a2t = (_Float16*)alloc((size_t)96 * CC * 2),
           *v1t = (_Float16*)alloc((size_t)64 * CC * 2),
           *v2t = (_Float16*)alloc((size_t)64 * CC * 2),
           *g1t = (_Float16*)alloc((size_t)256 * CC * 2),
           *g2t = (_Float16*)alloc((size_t)256 * CC * 2);
  _Float16 *hwh = (_Float16*)alloc((size_t)MM * 96 * 2),
           *hah = (_Float16*)alloc((size_t)MM * 96 * 2),
           *hvh = (_Float16*)alloc((size_t)MM * 64 * 2),
           *hgh = (_Float16*)alloc((size_t)MM * 256 * 2);
  float *rf  = (float*)alloc(F4), *kf = (float*)alloc(F4), *vf = (float*)alloc(F4),
        *decf = (float*)alloc(F4), *af = (float*)alloc(F4), *svf = (float*)alloc(F4),
        *aaf = (float*)alloc(F4), *bbf = (float*)alloc(F4),
        *of  = (float*)alloc(F4), *gf = (float*)alloc(F4);

  const dim3 blk(256);
  const int ecvt = (int)(MC / 256);

  // ---- weights -> f16 (transpose K x N -> N x K where needed) --------------
  cvt_f16<<<ecvt, blk, 0, stream>>>(W_r, Wrh, (int)MC);
  cvt_f16<<<ecvt, blk, 0, stream>>>(W_k, Wkh, (int)MC);
  cvt_f16<<<ecvt, blk, 0, stream>>>(W_v, Wvh, (int)MC);
  cvt_f16<<<ecvt, blk, 0, stream>>>(W_o, Woh, (int)MC);
  cvt_f16_t<<<(CC *  96 + 255) / 256, blk, 0, stream>>>(w1, w1t, CC,  96);
  cvt_f16_t<<<( 96 * CC + 255) / 256, blk, 0, stream>>>(w2, w2t,  96, CC);
  cvt_f16_t<<<(CC *  96 + 255) / 256, blk, 0, stream>>>(a1, a1t, CC,  96);
  cvt_f16_t<<<( 96 * CC + 255) / 256, blk, 0, stream>>>(a2, a2t,  96, CC);
  cvt_f16_t<<<(CC *  64 + 255) / 256, blk, 0, stream>>>(v1, v1t, CC,  64);
  cvt_f16_t<<<( 64 * CC + 255) / 256, blk, 0, stream>>>(v2, v2t,  64, CC);
  cvt_f16_t<<<(CC * 256 + 255) / 256, blk, 0, stream>>>(g1, g1t, CC, 256);
  cvt_f16_t<<<(256 * CC + 255) / 256, blk, 0, stream>>>(g2, g2t, 256, CC);

  // ---- time-shift + mixes --------------------------------------------------
  prep_mix<<<ecvt, blk, 0, stream>>>(x, x_r, x_w, x_k, x_v, x_a, x_g,
                                     xrh, xwh, xkh, xvh, xah, xgh);

  auto ggrid = [](int N_, int MT_, int NT_) {
    return (MM / (16 * MT_)) * (N_ / (16 * NT_)) / 8;
  };

  // ---- big projections (32x64 wave tiles) ----------------------------------
  gemm_wmma<2, 4, 0><<<ggrid(CC, 2, 4), blk, 0, stream>>>(xrh, Wrh, rf, nullptr, nullptr, MM, CC, CC);
  gemm_wmma<2, 4, 0><<<ggrid(CC, 2, 4), blk, 0, stream>>>(xkh, Wkh, kf, nullptr, nullptr, MM, CC, CC);
  gemm_wmma<2, 4, 0><<<ggrid(CC, 2, 4), blk, 0, stream>>>(xvh, Wvh, vf, nullptr, nullptr, MM, CC, CC);

  // ---- LoRA paths ----------------------------------------------------------
  gemm_wmma<2, 6, 2><<<ggrid( 96, 2, 6), blk, 0, stream>>>(xwh, w1t, nullptr, hwh, nullptr, MM,  96, CC);
  gemm_wmma<2, 4, 3><<<ggrid(CC, 2, 4), blk, 0, stream>>>(hwh, w2t, decf, nullptr, w0, MM, CC,  96);
  gemm_wmma<2, 4, 1><<<ggrid( 64, 2, 4), blk, 0, stream>>>(xvh, v1t, nullptr, hvh, nullptr, MM,  64, CC);
  gemm_wmma<2, 4, 4><<<ggrid(CC, 2, 4), blk, 0, stream>>>(hvh, v2t, svf, nullptr, v0, MM, CC,  64);
  gemm_wmma<2, 6, 1><<<ggrid( 96, 2, 6), blk, 0, stream>>>(xah, a1t, nullptr, hah, nullptr, MM,  96, CC);
  gemm_wmma<2, 4, 4><<<ggrid(CC, 2, 4), blk, 0, stream>>>(hah, a2t, af, nullptr, a0, MM, CC,  96);
  gemm_wmma<2, 4, 5><<<ggrid(256, 2, 4), blk, 0, stream>>>(xgh, g1t, nullptr, hgh, nullptr, MM, 256, CC);
  gemm_wmma<2, 4, 0><<<ggrid(CC, 2, 4), blk, 0, stream>>>(hgh, g2t, gf, nullptr, nullptr, MM, CC, 256);

  // ---- per-head pre-scan ---------------------------------------------------
  head_mix<<<(MM * HH) / 8, blk, 0, stream>>>(af, svf, v_first, kf, vf, aaf, bbf, k_k, k_a);

  // ---- sequential WKV7 scan ------------------------------------------------
  wkv_scan<<<BB * HH, dim3(64), 0, stream>>>(rf, decf, kf, vf, aaf, bbf, of);

  // ---- per-head post-scan (GroupNorm + residual + gate) -> f16 -------------
  post_mix<<<(MM * HH) / 8, blk, 0, stream>>>(of, rf, kf, vf, gf, r_k, ln_w, ln_b, ogh);

  // ---- output projection straight into d_out -------------------------------
  gemm_wmma<2, 4, 0><<<ggrid(CC, 2, 4), blk, 0, stream>>>(ogh, Woh, outp, nullptr, nullptr, MM, CC, CC);

  // ---- second tuple output: v_first passthrough ----------------------------
  hipMemcpyAsync(outp + MC, v_first, MC * sizeof(float), hipMemcpyDeviceToDevice, stream);
}
